// GInRec_32341103739245
// MI455X (gfx1250) — matrix operand order repaired and checked by reading, not compiled, not run on hardware
//
#include <hip/hip_runtime.h>
#include <hip/hip_bf16.h>
#include <math.h>

typedef __attribute__((ext_vector_type(16))) _Float16 v16h;
typedef __attribute__((ext_vector_type(8)))  float    v8f;
typedef __attribute__((ext_vector_type(8)))  unsigned v8u;

#define N_NODES 50000
#define N_EDGES 800000
#define N_REL   8
#define ENT_DIM 384
#define AE0     128
#define DDIM    64
#define ATT_DIM 32
#define NEG_SLOPE 0.01f

#ifndef NO_TDM
#if defined(__has_builtin)
#if __has_builtin(__builtin_amdgcn_tensor_load_to_lds)
#define USE_TDM 1
#endif
#endif
#endif

#ifdef USE_TDM
typedef __attribute__((ext_vector_type(4))) unsigned tdm_v4u;
typedef __attribute__((ext_vector_type(8))) int      tdm_v8i;
typedef __attribute__((ext_vector_type(4))) int      tdm_v4i;
#endif

// ---------- WMMA helpers (CDNA5 gfx1250, wave32) ----------

static __device__ inline v8f wmma_f16(v16h a, v16h b, v8f c) {
    return __builtin_amdgcn_wmma_f32_16x16x32_f16(
        false, a, false, b, (short)0, c, false, false);
}

// A-matrix (16x32 f16) layout: lane L holds row m=L%16.
// VGPR v (pair j=2v,2v+1): k = (v<4 ? 2v : 8+2v) + 8*(L/16) + (j&1)
static __device__ inline int a_kloc(int j, int half) {
    int v = j >> 1, p = j & 1;
    return ((v < 4) ? 2 * v : 8 + 2 * v) + 8 * half + p;
}
// B-matrix (32x16 f16) layout: lane L holds col n=L%16.
// VGPR v: k = (v<4 ? 2v : 8+2v) + 8*(L/16) + p
static __device__ inline int b_kloc(int j, int half) {
    int v = j >> 1, p = j & 1;
    return ((v < 4) ? 2 * v : 8 + 2 * v) + 8 * half + p;
}

// Load one A fragment from a row-major f16 row base (k contiguous), using
// packed 32-bit pair loads (each VGPR pair is two consecutive k).
static __device__ inline v16h load_a_pairs(const _Float16* rowbase, int half) {
    v8u t;
#pragma unroll
    for (int v = 0; v < 8; ++v) {
        int kb = ((v < 4) ? 2 * v : 8 + 2 * v) + 8 * half;
        t[v] = *(const unsigned*)(rowbase + kb);
    }
    return __builtin_bit_cast(v16h, t);
}

// Build one A fragment from an f32 row base (converting to f16).
static __device__ inline v16h cvt_a_frag_f32(const float* rowbase, int half) {
    v16h a;
#pragma unroll
    for (int v = 0; v < 8; ++v) {
        int kb = ((v < 4) ? 2 * v : 8 + 2 * v) + 8 * half;
        float2 p = *(const float2*)(rowbase + kb);
        a[2 * v]     = (_Float16)p.x;
        a[2 * v + 1] = (_Float16)p.y;
    }
    return a;
}

// Load a pre-packed B fragment (512 halves per tile; lane-contiguous 32B)
static __device__ inline v16h load_b_frag(const _Float16* packed, int numN,
                                          int kt, int nt, int lane) {
    return *(const v16h*)(packed + ((size_t)(kt * numN + nt) * 512) + lane * 16);
}

static __device__ inline float lrelu(float v) {
    return v > 0.f ? v : NEG_SLOPE * v;
}

static __device__ inline float atomicMaxF(float* addr, float val) {
    unsigned* ua = (unsigned*)addr;
    unsigned old = *ua;
    while (__uint_as_float(old) < val) {
        unsigned assumed = old;
        old = atomicCAS(ua, assumed, __float_as_uint(val));
        if (old == assumed) break;
    }
    return __uint_as_float(old);
}

// ---------- kernel 0: pack a KxN f32 matrix into WMMA-B fragment order ----------
__global__ void pack_b_kernel(const float* __restrict__ W, _Float16* __restrict__ out,
                              int K, int Nn) {
    int total = (K / 32) * (Nn / 16) * 512;
    int idx = blockIdx.x * blockDim.x + threadIdx.x;
    if (idx >= total) return;
    int j    = idx & 15;
    int lane = (idx >> 4) & 31;
    int tile = idx >> 9;
    int numN = Nn / 16;
    int nt = tile % numN;
    int kt = tile / numN;
    int half = lane >> 4;
    int k = kt * 32 + b_kloc(j, half);
    int n = nt * 16 + (lane & 15);
    out[idx] = (_Float16)W[(size_t)k * Nn + n];
}

// ---------- kernel 1: encoder  h = (x@W0+b0)@W1+b1 ----------
// 2 waves/block; each wave stages its 16x384 x-tile into LDS (TDM if
// available, else manually coalesced float4 loads), then runs the WMMA chain.
__global__ __launch_bounds__(64) void encode_kernel(
    const float* __restrict__ x, const float* __restrict__ b0v,
    const float* __restrict__ b1v, const _Float16* __restrict__ w0p,
    const _Float16* __restrict__ w1p, float* __restrict__ hf,
    _Float16* __restrict__ hh) {
    __shared__ float    xs[2][16 * ENT_DIM];   // 48KB: per-wave x tile (f32)
    __shared__ _Float16 ts[2][16 * AE0];       // 8KB:  per-wave t staging (f16)
    int lane = threadIdx.x & 31;
    int wave = __builtin_amdgcn_readfirstlane(threadIdx.x >> 5);
    int mtile = blockIdx.x * 2 + wave;
    if (mtile >= N_NODES / 16) return;        // wave-uniform exit
    int row0 = mtile * 16;
    int half = lane >> 4;
    int m    = lane & 15;

#ifdef USE_TDM
    {
        // Tensor DMA: 2D tile (16 rows x 384 f32, row stride 384) -> LDS
        unsigned ldsoff = (unsigned)(uintptr_t)&xs[wave][0];
        unsigned long long ga =
            (unsigned long long)(uintptr_t)(x + (size_t)row0 * ENT_DIM);
        tdm_v4u g0;
        g0[0] = 1u;                                  // count=1, user mode
        g0[1] = ldsoff;                              // lds_addr
        g0[2] = (unsigned)(ga & 0xFFFFFFFFu);        // global_addr[31:0]
        g0[3] = (unsigned)((ga >> 32) & 0x1FFFFFFu)  // global_addr[56:32]
                | (2u << 30);                        // type=2 (image)
        tdm_v8i g1;
        g1[0] = (int)(2u << 16);                     // data_size=4B
        g1[1] = (int)((ENT_DIM & 0xFFFFu) << 16);    // tensor_dim0[15:0]
        g1[2] = (int)(((unsigned)ENT_DIM >> 16) |
                      (((unsigned)N_NODES & 0xFFFFu) << 16)); // dim0 hi, dim1 lo
        g1[3] = (int)((((unsigned)N_NODES >> 16) & 0xFFFFu) |
                      ((unsigned)ENT_DIM << 16));    // dim1 hi, tile_dim0=384
        g1[4] = 16;                                  // tile_dim1=16, tile_dim2=0
        g1[5] = ENT_DIM;                             // tensor_dim0_stride lo32
        g1[6] = 0;
        g1[7] = 0;
        tdm_v4i g2 = {0, 0, 0, 0};
        tdm_v4i g3 = {0, 0, 0, 0};
        tdm_v8i g4 = {0, 0, 0, 0, 0, 0, 0, 0};
        __builtin_amdgcn_tensor_load_to_lds(g0, g1, g2, g3, g4, 0);
        __builtin_amdgcn_s_wait_tensorcnt(0);
    }
#else
    {
        // Manually coalesced staging: lanes cover contiguous 128B row chunks
        const float* xbase = x + (size_t)row0 * ENT_DIM;
#pragma unroll
        for (int it = 0; it < 48; ++it) {
            int f4 = it * 32 + lane;              // float4 index in 16x384 tile
            int rr = f4 / 96;                     // row (96 float4 per row)
            int cc = (f4 % 96) * 4;               // col
            float4 v = *(const float4*)(xbase + (size_t)rr * ENT_DIM + cc);
            *(float4*)&xs[wave][rr * ENT_DIM + cc] = v;
        }
    }
#endif

    // Build all 12 A fragments from the LDS-staged tile (f32 -> f16)
    v16h afr[12];
#pragma unroll
    for (int kt = 0; kt < 12; ++kt)
        afr[kt] = cvt_a_frag_f32(&xs[wave][m * ENT_DIM + kt * 32], half);

    // Stage 1: t[16,128] = x@W0 + b0 -> LDS (f16); B loads software-pipelined
    for (int nt = 0; nt < 8; ++nt) {
        v8f acc = {};
        v16h bcur = load_b_frag(w0p, 8, 0, nt, lane);
#pragma unroll
        for (int kt = 0; kt < 12; ++kt) {
            v16h bnext = (kt < 11) ? load_b_frag(w0p, 8, kt + 1, nt, lane) : bcur;
            acc = wmma_f16(afr[kt], bcur, acc);
            bcur = bnext;
        }
        float bias = b0v[nt * 16 + m];
#pragma unroll
        for (int i = 0; i < 8; ++i)
            ts[wave][(i + 8 * half) * AE0 + nt * 16 + m] = (_Float16)(acc[i] + bias);
    }
    // Stage 2: h[16,64] = t@W1 + b1
    v16h a2[4];
#pragma unroll
    for (int kt = 0; kt < 4; ++kt)
        a2[kt] = load_a_pairs(&ts[wave][m * AE0 + kt * 32], half);
    for (int nt = 0; nt < 4; ++nt) {
        v8f acc = {};
        v16h bcur = load_b_frag(w1p, 4, 0, nt, lane);
#pragma unroll
        for (int kt = 0; kt < 4; ++kt) {
            v16h bnext = (kt < 3) ? load_b_frag(w1p, 4, kt + 1, nt, lane) : bcur;
            acc = wmma_f16(a2[kt], bcur, acc);
            bcur = bnext;
        }
        float bias = b1v[nt * 16 + m];
#pragma unroll
        for (int i = 0; i < 8; ++i) {
            int mm = i + 8 * half;
            int nn = nt * 16 + m;
            float v = acc[i] + bias;
            size_t gi = (size_t)(row0 + mm) * DDIM + nn;
            hf[gi] = v;
            hh[gi] = (_Float16)v;
        }
    }
}

// ---------- relation bucketing ----------
__global__ void hist_kernel(const int* __restrict__ et, int* __restrict__ counts) {
    int e = blockIdx.x * blockDim.x + threadIdx.x;
    if (e < N_EDGES) atomicAdd(&counts[et[e]], 1);
}

__global__ void offsets_kernel(const int* __restrict__ counts, int* offs,
                               int* cursors, int* tile_offs) {
    if (threadIdx.x == 0 && blockIdx.x == 0) {
        int acc = 0, tacc = 0;
        for (int r = 0; r < N_REL; ++r) {
            offs[r] = acc; cursors[r] = acc; tile_offs[r] = tacc;
            tacc += (counts[r] + 15) / 16;
            acc  += counts[r];
        }
        offs[N_REL] = acc; tile_offs[N_REL] = tacc;
    }
}

__global__ void scatter_kernel(const int* __restrict__ et, int* __restrict__ cursors,
                               int* __restrict__ perm) {
    int e = blockIdx.x * blockDim.x + threadIdx.x;
    if (e < N_EDGES) {
        int pos = atomicAdd(&cursors[et[e]], 1);
        perm[pos] = e;
    }
}

// ---------- pass 1: attention scores (relation independent: alpha[0]) ----------
// Persistent waves: all W_a B-fragments live in registers; grid-stride loop.
__global__ __launch_bounds__(256) void score_kernel(
    const _Float16* __restrict__ hh, const int* __restrict__ esrc,
    const int* __restrict__ edst, const _Float16* __restrict__ wap,
    const float* __restrict__ b_a, const float* __restrict__ alpha0,
    float* __restrict__ score) {
    int lane = threadIdx.x & 31;
    int wave = threadIdx.x >> 5;
    int wid  = blockIdx.x * 8 + wave;
    int nwaves = gridDim.x * 8;
    int half = lane >> 4;
    int m = lane & 15;

    // resident B fragments: 2 N-tiles x 4 K-tiles
    v16h bfr[2][4];
#pragma unroll
    for (int nt = 0; nt < 2; ++nt)
#pragma unroll
        for (int kt = 0; kt < 4; ++kt)
            bfr[nt][kt] = load_b_frag(wap, 2, kt, nt, lane);
    float bias[2], al[2];
#pragma unroll
    for (int nt = 0; nt < 2; ++nt) {
        bias[nt] = b_a[nt * 16 + m];
        al[nt]   = alpha0[nt * 16 + m];
    }

    for (int t = wid; t < N_EDGES / 16; t += nwaves) {
        int e = t * 16 + m;
        int vd = edst[e], vs = esrc[e];

        v16h afr[4];
#pragma unroll
        for (int kt = 0; kt < 2; ++kt)
            afr[kt] = load_a_pairs(hh + (size_t)vd * DDIM + kt * 32, half);
#pragma unroll
        for (int kt = 0; kt < 2; ++kt)
            afr[2 + kt] = load_a_pairs(hh + (size_t)vs * DDIM + kt * 32, half);

        float part[8];
#pragma unroll
        for (int i = 0; i < 8; ++i) part[i] = 0.f;

#pragma unroll
        for (int nt = 0; nt < 2; ++nt) {
            v8f acc = {};
#pragma unroll
            for (int kt = 0; kt < 4; ++kt)
                acc = wmma_f16(afr[kt], bfr[nt][kt], acc);
#pragma unroll
            for (int i = 0; i < 8; ++i)
                part[i] += lrelu(acc[i] + bias[nt]) * al[nt];
        }
        // reduce over the 16 columns held by each half-wave
#pragma unroll
        for (int i = 0; i < 8; ++i) {
            float v = part[i];
            v += __shfl_xor(v, 1);
            v += __shfl_xor(v, 2);
            v += __shfl_xor(v, 4);
            v += __shfl_xor(v, 8);
            if (m == i)
                score[t * 16 + i + 8 * half] = v;
        }
    }
}

// ---------- segment softmax ----------
__global__ void init_neginf_kernel(float* __restrict__ p, int n) {
    int i = blockIdx.x * blockDim.x + threadIdx.x;
    if (i < n) p[i] = -INFINITY;
}

__global__ void segmax_kernel(const float* __restrict__ score,
                              const int* __restrict__ edst, float* __restrict__ marr) {
    int e = blockIdx.x * blockDim.x + threadIdx.x;
    if (e < N_EDGES) atomicMaxF(&marr[edst[e]], score[e]);
}

__global__ void expsum_kernel(const float* __restrict__ score,
                              const int* __restrict__ edst,
                              const float* __restrict__ marr,
                              float* __restrict__ escore, float* __restrict__ sarr) {
    int e = blockIdx.x * blockDim.x + threadIdx.x;
    if (e < N_EDGES) {
        float ex = __expf(score[e] - marr[edst[e]]);
        escore[e] = ex;
        atomicAdd(&sarr[edst[e]], ex);
    }
}

// ---------- pass 2: relation-gated GEMM + weighted aggregation ----------
// Chunked persistent waves: current relation's W_r fragments cached in regs,
// reloaded only when the relation changes within a wave's contiguous chunk.
__global__ __launch_bounds__(256) void gate_agg_kernel(
    const _Float16* __restrict__ hh, const float* __restrict__ hf,
    const int* __restrict__ esrc, const int* __restrict__ edst,
    const int* __restrict__ perm, const int* __restrict__ offs,
    const int* __restrict__ tile_offs, const _Float16* __restrict__ wrp,
    const float* __restrict__ escore, const float* __restrict__ sarr,
    float* __restrict__ hN) {
    int lane = threadIdx.x & 31;
    int wave = threadIdx.x >> 5;
    int wid  = blockIdx.x * 8 + wave;
    int nwaves = gridDim.x * 8;
    int half = lane >> 4;
    int m = lane & 15;

    int total = tile_offs[N_REL];
    int chunk = (total + nwaves - 1) / nwaves;
    int gt0 = wid * chunk;
    int gt1 = gt0 + chunk;
    if (gt1 > total) gt1 = total;

    v16h bfr[4][4];
    int cur_r = -1;

    for (int gt = gt0; gt < gt1; ++gt) {
        int r = 0;
        while (r < N_REL - 1 && gt >= tile_offs[r + 1]) ++r;
        if (r != cur_r) {
            const _Float16* wrp_r = wrp + (size_t)r * 16 * 512;
            __builtin_prefetch(wrp_r, 0, 3);
#pragma unroll
            for (int nt = 0; nt < 4; ++nt)
#pragma unroll
                for (int kt = 0; kt < 4; ++kt)
                    bfr[nt][kt] = load_b_frag(wrp_r, 4, kt, nt, lane);
            cur_r = r;
        }
        int lt = gt - tile_offs[r];
        int ebase  = offs[r] + lt * 16;
        int ecount = offs[r + 1] - ebase;
        if (ecount > 16) ecount = 16;

        int mrow = (m < ecount) ? m : (ecount - 1);
        int e  = perm[ebase + mrow];
        int vd = edst[e], vs = esrc[e];
        float att_l = escore[e] / sarr[vd];

        v16h afr[4];
#pragma unroll
        for (int kt = 0; kt < 2; ++kt)
            afr[kt] = load_a_pairs(hh + (size_t)vd * DDIM + kt * 32, half);
#pragma unroll
        for (int kt = 0; kt < 2; ++kt)
            afr[2 + kt] = load_a_pairs(hh + (size_t)vs * DDIM + kt * 32, half);

#pragma unroll
        for (int nt = 0; nt < 4; ++nt) {
            v8f acc = {};
#pragma unroll
            for (int kt = 0; kt < 4; ++kt)
                acc = wmma_f16(afr[kt], bfr[nt][kt], acc);
            int nn = nt * 16 + m;
#pragma unroll
            for (int i = 0; i < 8; ++i) {
                int mm = i + 8 * half;
                int sl = i + 8 * half;      // lane holding row mm state
                float att_row = __shfl(att_l, sl);
                int srcrow = __shfl(vs, sl);
                int dstrow = __shfl(vd, sl);
                if (mm < ecount) {
                    float gate = 1.f / (1.f + __expf(-acc[i]));
                    float msg = hf[(size_t)srcrow * DDIM + nn] * att_row * gate;
                    atomicAdd(&hN[(size_t)dstrow * DDIM + nn], msg);
                }
            }
        }
    }
}

// ---------- final: out = leakyrelu(hN @ lin_w + lin_b) ----------
__global__ __launch_bounds__(256) void final_kernel(
    const float* __restrict__ hN, const _Float16* __restrict__ linp,
    const float* __restrict__ lin_b, float* __restrict__ out) {
    int lane = threadIdx.x & 31;
    int wave = threadIdx.x >> 5;
    int wid  = blockIdx.x * 8 + wave;
    int nwaves = gridDim.x * 8;
    int half = lane >> 4;
    int m = lane & 15;

    // resident B fragments: 4 N-tiles x 2 K-tiles
    v16h bfr[4][2];
#pragma unroll
    for (int nt = 0; nt < 4; ++nt)
#pragma unroll
        for (int kt = 0; kt < 2; ++kt)
            bfr[nt][kt] = load_b_frag(linp, 4, kt, nt, lane);

    for (int mtile = wid; mtile < N_NODES / 16; mtile += nwaves) {
        int row0 = mtile * 16;
        v16h afr[2];
        const float* hrow = hN + (size_t)(row0 + m) * DDIM;
#pragma unroll
        for (int kt = 0; kt < 2; ++kt)
            afr[kt] = cvt_a_frag_f32(hrow + kt * 32, half);
#pragma unroll
        for (int nt = 0; nt < 4; ++nt) {
            v8f acc = {};
#pragma unroll
            for (int kt = 0; kt < 2; ++kt)
                acc = wmma_f16(afr[kt], bfr[nt][kt], acc);
            int nn = nt * 16 + m;
            float bias = lin_b[nn];
#pragma unroll
            for (int i = 0; i < 8; ++i) {
                int mm = i + 8 * half;
                out[(size_t)(row0 + mm) * DDIM + nn] = lrelu(acc[i] + bias);
            }
        }
    }
}

// ---------- host-side orchestration ----------
static inline size_t align_up(size_t v, size_t a) { return (v + a - 1) & ~(a - 1); }

extern "C" void kernel_launch(void* const* d_in, const int* in_sizes, int n_in,
                              void* d_out, int out_size, void* d_ws, size_t ws_size,
                              hipStream_t stream) {
    (void)in_sizes; (void)n_in; (void)out_size; (void)ws_size;
    const float* x      = (const float*)d_in[0];
    const int*   esrc   = (const int*)d_in[1];
    const int*   edst   = (const int*)d_in[2];
    const int*   etype  = (const int*)d_in[3];
    const float* enc_w0 = (const float*)d_in[4];
    const float* enc_b0 = (const float*)d_in[5];
    const float* enc_w1 = (const float*)d_in[6];
    const float* enc_b1 = (const float*)d_in[7];
    const float* W_r    = (const float*)d_in[8];
    const float* W_a    = (const float*)d_in[9];
    const float* b_a    = (const float*)d_in[10];
    const float* alpha  = (const float*)d_in[11];
    const float* lin_w  = (const float*)d_in[12];
    const float* lin_b  = (const float*)d_in[13];
    float* out = (float*)d_out;

    // workspace carve-up
    char* w = (char*)d_ws;
    size_t off = 0;
    auto take = [&](size_t bytes) {
        void* p = w + off;
        off = align_up(off + bytes, 256);
        return p;
    };
    float*     hf     = (float*)take((size_t)N_NODES * DDIM * 4);
    _Float16*  hh     = (_Float16*)take((size_t)N_NODES * DDIM * 2);
    float*     score  = (float*)take((size_t)N_EDGES * 4);
    float*     escore = (float*)take((size_t)N_EDGES * 4);
    float*     marr   = (float*)take((size_t)N_NODES * 4);
    float*     sarr   = (float*)take((size_t)N_NODES * 4);
    float*     hN     = (float*)take((size_t)N_NODES * DDIM * 4);
    int*       perm   = (int*)take((size_t)N_EDGES * 4);
    _Float16*  w0p    = (_Float16*)take((size_t)96 * 512 * 2);   // 384x128
    _Float16*  w1p    = (_Float16*)take((size_t)16 * 512 * 2);   // 128x64
    _Float16*  wap    = (_Float16*)take((size_t)8  * 512 * 2);   // 128x32
    _Float16*  wrp    = (_Float16*)take((size_t)N_REL * 16 * 512 * 2); // 8x(128x64)
    _Float16*  linp   = (_Float16*)take((size_t)8  * 512 * 2);   // 64x64
    int*       counts   = (int*)take(N_REL * 4);
    int*       offs     = (int*)take((N_REL + 1) * 4);
    int*       cursors  = (int*)take(N_REL * 4);
    int*       tileoffs = (int*)take((N_REL + 1) * 4);

    // zero init (capture-safe async memsets)
    (void)hipMemsetAsync(sarr, 0, (size_t)N_NODES * 4, stream);
    (void)hipMemsetAsync(hN, 0, (size_t)N_NODES * DDIM * 4, stream);
    (void)hipMemsetAsync(counts, 0, N_REL * 4, stream);
    init_neginf_kernel<<<(N_NODES + 255) / 256, 256, 0, stream>>>(marr, N_NODES);

    // pack weights into WMMA-B fragment order
    pack_b_kernel<<<(96 * 512 + 255) / 256, 256, 0, stream>>>(enc_w0, w0p, ENT_DIM, AE0);
    pack_b_kernel<<<(16 * 512 + 255) / 256, 256, 0, stream>>>(enc_w1, w1p, AE0, DDIM);
    pack_b_kernel<<<(8 * 512 + 255) / 256, 256, 0, stream>>>(W_a, wap, 2 * DDIM, ATT_DIM);
    for (int r = 0; r < N_REL; ++r)
        pack_b_kernel<<<(16 * 512 + 255) / 256, 256, 0, stream>>>(
            W_r + (size_t)r * 2 * DDIM * DDIM, wrp + (size_t)r * 16 * 512,
            2 * DDIM, DDIM);
    pack_b_kernel<<<(8 * 512 + 255) / 256, 256, 0, stream>>>(lin_w, linp, DDIM, DDIM);

    // encoder: 3125 M-tiles, 2 waves/block (56KB LDS/block)
    encode_kernel<<<1563, 64, 0, stream>>>(x, enc_b0, enc_b1, w0p, w1p, hf, hh);

    // relation bucketing
    hist_kernel<<<N_EDGES / 256, 256, 0, stream>>>(etype, counts);
    offsets_kernel<<<1, 32, 0, stream>>>(counts, offs, cursors, tileoffs);
    scatter_kernel<<<N_EDGES / 256, 256, 0, stream>>>(etype, cursors, perm);

    // attention scores: persistent waves, W_a resident in registers
    score_kernel<<<256, 256, 0, stream>>>(hh, esrc, edst, wap, b_a, alpha, score);

    // edge softmax
    segmax_kernel<<<N_EDGES / 256, 256, 0, stream>>>(score, edst, marr);
    expsum_kernel<<<N_EDGES / 256, 256, 0, stream>>>(score, edst, marr, escore, sarr);

    // gated message + aggregation: chunked persistent waves, W_r cached in regs
    gate_agg_kernel<<<256, 256, 0, stream>>>(hh, hf, esrc, edst, perm, offs,
                                             tileoffs, wrp, escore, sarr, hN);

    // final linear + leaky relu: persistent waves, lin_w resident
    final_kernel<<<256, 256, 0, stream>>>(hN, linp, lin_b, out);
}
